// WordLSTM_21105469292870
// MI455X (gfx1250) — compile-verified, matrix operand order checked by reading, not run on hardware
//
#include <hip/hip_runtime.h>
#include <hip/hip_bf16.h>

typedef __bf16 bf16;
typedef __attribute__((ext_vector_type(8)))  __bf16 v8bf;
typedef __attribute__((ext_vector_type(16))) __bf16 v16bf;
typedef __attribute__((ext_vector_type(8)))  float  v8f;

#define BB 32
#define SS 512
#define EE 512
#define HH 1024
#define G4 4096   // 4*HH

// ---------------------------------------------------------------------------
// WMMA helpers (CDNA5, wave32, v_wmma_f32_16x16x32_bf16)
// A/B 16-bit fragment layout (ISA 7.12.2): lane l holds row (l&15);
// lanes 0-15: K = {0..7, 16..23}; lanes 16-31: K = {8..15, 24..31}.
// ---------------------------------------------------------------------------
__device__ __forceinline__ v16bf load_frag(const bf16* __restrict__ base, int ld) {
  int l  = threadIdx.x & 31;
  int r  = l & 15;
  int kb = (l >> 4) << 3;              // 0 or 8
  const bf16* p = base + (size_t)r * ld + kb;
  v8bf lo = *(const v8bf*)(p);         // K = kb .. kb+7
  v8bf hi = *(const v8bf*)(p + 16);    // K = kb+16 .. kb+23
  return __builtin_shufflevector(lo, hi, 0,1,2,3,4,5,6,7,8,9,10,11,12,13,14,15);
}

__device__ __forceinline__ v8f wmma_bf16(v16bf a, v16bf b, v8f c) {
  return __builtin_amdgcn_wmma_f32_16x16x32_bf16(false, a, false, b, (short)0, c,
                                                 false, false);
}

__device__ __forceinline__ float sigf(float x) { return 1.0f / (1.0f + __expf(-x)); }

// ---------------------------------------------------------------------------
// Elementwise utility kernels
// ---------------------------------------------------------------------------
__global__ __launch_bounds__(256) void f2bf_kernel(const float* __restrict__ s,
                                                   bf16* __restrict__ d, size_t n) {
  size_t stride = (size_t)gridDim.x * blockDim.x;
  for (size_t i = (size_t)blockIdx.x * blockDim.x + threadIdx.x; i < n; i += stride)
    d[i] = (bf16)s[i];
}

__global__ __launch_bounds__(256) void zero_f32_kernel(float* __restrict__ d, size_t n) {
  size_t stride = (size_t)gridDim.x * blockDim.x;
  for (size_t i = (size_t)blockIdx.x * blockDim.x + threadIdx.x; i < n; i += stride)
    d[i] = 0.0f;
}

__global__ __launch_bounds__(256) void zero_bf_kernel(bf16* __restrict__ d, size_t n) {
  size_t stride = (size_t)gridDim.x * blockDim.x;
  for (size_t i = (size_t)blockIdx.x * blockDim.x + threadIdx.x; i < n; i += stride)
    d[i] = (bf16)0.0f;
}

// x (B,S,E) fp32 -> xt (S,B,E): fp32 copy into out accumulator (residual init
// for layer 0) and bf16 copy (GEMM A operand).
__global__ __launch_bounds__(256) void transpose_x_kernel(const float* __restrict__ x,
                                                          float* __restrict__ outb,
                                                          bf16* __restrict__ xtb) {
  size_t n = (size_t)BB * SS * EE;
  size_t stride = (size_t)gridDim.x * blockDim.x;
  for (size_t i = (size_t)blockIdx.x * blockDim.x + threadIdx.x; i < n; i += stride) {
    size_t e = i % EE;
    size_t s = (i / EE) % SS;
    size_t b = i / ((size_t)EE * SS);
    float v = x[i];
    size_t o = (s * BB + b) * EE + e;
    outb[o] = v;
    xtb[o]  = (bf16)v;
  }
}

// (S,B,E) fp32 -> (B,S,E) fp32 (final output)
__global__ __launch_bounds__(256) void final_transpose_kernel(const float* __restrict__ outb,
                                                              float* __restrict__ out) {
  size_t n = (size_t)BB * SS * EE;
  size_t stride = (size_t)gridDim.x * blockDim.x;
  for (size_t i = (size_t)blockIdx.x * blockDim.x + threadIdx.x; i < n; i += stride) {
    size_t e = i % EE;
    size_t s = (i / EE) % SS;
    size_t b = i / ((size_t)EE * SS);
    out[i] = outb[(s * BB + b) * EE + e];
  }
}

// ---------------------------------------------------------------------------
// K1: input-to-gates GEMM.  G(16384x4096) = A(16384x512,bf16) @ W^T + bias
// W stored row-major (4096 x 512): output column j uses W row j.
// One wave -> 64x64 output tile (4x4 WMMA tiles): 16 WMMAs per 16 b128 loads
// per K-step (1 load/WMMA), K loop of 16.
// grid = 2048 blocks x 256 threads = 16384 waves = (256 M-tiles)x(64 N-tiles)
// ---------------------------------------------------------------------------
__global__ __launch_bounds__(256) void gemm_gates_kernel(const bf16* __restrict__ A,
                                                         const bf16* __restrict__ W,
                                                         const float* __restrict__ bias,
                                                         float* __restrict__ G) {
  const int NT = G4 / 64;  // 64 column tiles
  int wave = blockIdx.x * (blockDim.x >> 5) + (threadIdx.x >> 5);
  int wm = wave / NT;      // 0..255
  int wn = wave % NT;      // 0..63
  int m0 = wm * 64, n0 = wn * 64;

  v8f acc[4][4] = {};
  for (int k0 = 0; k0 < EE; k0 += 32) {
    v16bf a[4], b[4];
    #pragma unroll
    for (int i = 0; i < 4; i++) {
      a[i] = load_frag(A + (size_t)(m0 + 16 * i) * EE + k0, EE);
      b[i] = load_frag(W + (size_t)(n0 + 16 * i) * EE + k0, EE);
    }
    #pragma unroll
    for (int mi = 0; mi < 4; mi++)
      #pragma unroll
      for (int ni = 0; ni < 4; ni++)
        acc[mi][ni] = wmma_bf16(a[mi], b[ni], acc[mi][ni]);
  }

  // C/D layout: lane l holds col (l&15), rows (l>=16 ? 8..15 : 0..7) in v[0..7]
  int l = threadIdx.x & 31;
  int n = l & 15, mb = (l >> 4) << 3;
  #pragma unroll
  for (int mi = 0; mi < 4; mi++)
    #pragma unroll
    for (int ni = 0; ni < 4; ni++) {
      int nc = n0 + ni * 16 + n;
      float bv = bias[nc];
      float* o = G + (size_t)(m0 + mi * 16 + mb) * G4 + nc;
      #pragma unroll
      for (int v = 0; v < 8; v++) o[(size_t)v * G4] = acc[mi][ni][v] + bv;
    }
}

// ---------------------------------------------------------------------------
// K2: one recurrent step, both directions: gates = G[t] + hp @ Whh^T, then
// cell update.  grid = (32 hidden tiles, 2 dirs), 256 threads (8 waves).
// Wave w handles gate q=w>>1, sub-tile jt=w&1 -> 16 gate columns, full M=32.
// i/f/g/o exchanged via LDS for the pointwise cell update.
// ---------------------------------------------------------------------------
__global__ __launch_bounds__(256) void lstm_gate_step_kernel(const bf16* __restrict__ hp,   // [2][32][512]
                                                             const bf16* __restrict__ Whh,  // [2][4096][512] (layer base)
                                                             const float* __restrict__ G,   // [2][S][B][4096]
                                                             float* __restrict__ c,         // [2][32][1024]
                                                             bf16* __restrict__ hbuf,       // [2][32][1024]
                                                             int t) {
  __shared__ float lds[4][BB][32];
  int dir = blockIdx.y;
  int j0  = blockIdx.x * 32;
  int tg  = dir ? (SS - 1 - t) : t;

  const bf16*  A = hp  + (size_t)dir * BB * EE;
  const bf16*  W = Whh + (size_t)dir * G4 * EE;
  const float* g = G   + ((size_t)dir * SS + tg) * ((size_t)BB * G4);

  int w  = threadIdx.x >> 5;     // 0..7
  int q  = w >> 1;               // gate index: i,f,g,o
  int jt = w & 1;
  int col0 = q * HH + j0 + jt * 16;   // W_hh row base / gate column base

  v8f acc0 = {}, acc1 = {};
  for (int k0 = 0; k0 < EE; k0 += 32) {
    v16bf b  = load_frag(W + (size_t)col0 * EE + k0, EE);
    v16bf a0 = load_frag(A + k0, EE);
    v16bf a1 = load_frag(A + (size_t)16 * EE + k0, EE);
    acc0 = wmma_bf16(a0, b, acc0);
    acc1 = wmma_bf16(a1, b, acc1);
  }

  int l = threadIdx.x & 31;
  int n = l & 15, mb = (l >> 4) << 3;
  #pragma unroll
  for (int v = 0; v < 8; v++) {
    int m = mb + v;
    lds[q][m][jt * 16 + n]      = acc0[v] + g[(size_t)m * G4 + col0 + n];
    lds[q][m + 16][jt * 16 + n] = acc1[v] + g[(size_t)(m + 16) * G4 + col0 + n];
  }
  __syncthreads();

  for (int e = threadIdx.x; e < BB * 32; e += blockDim.x) {
    int b_ = e >> 5, jj = e & 31;
    int j = j0 + jj;
    float iv = lds[0][b_][jj], fv = lds[1][b_][jj];
    float gv = lds[2][b_][jj], ov = lds[3][b_][jj];
    size_t ci = (size_t)dir * BB * HH + (size_t)b_ * HH + j;
    float cn = sigf(fv) * c[ci] + sigf(iv) * tanhf(gv);
    float hv = sigf(ov) * tanhf(cn);
    c[ci]    = cn;
    hbuf[ci] = (bf16)hv;
  }
}

// ---------------------------------------------------------------------------
// K3: projection hp = h @ Whr^T (M=32, N=512, K=1024), both dirs; writes
// bf16 hp for the next step and accumulates the fp32 layer output.
// grid = (8 N-tiles of 64, 2 dirs), 8 waves: wave w -> (mi=w&1, 16-col tile w>>1)
// ---------------------------------------------------------------------------
__global__ __launch_bounds__(256) void proj_step_kernel(const bf16* __restrict__ hbuf, // [2][32][1024]
                                                        const bf16* __restrict__ Whr,  // [2][512][1024] (layer base)
                                                        bf16* __restrict__ hp,         // [2][32][512]
                                                        float* __restrict__ outb,      // [S][B][E]
                                                        int t) {
  int dir  = blockIdx.y;
  int sOut = dir ? (SS - 1 - t) : t;
  const bf16* A = hbuf + (size_t)dir * BB * HH;
  const bf16* W = Whr  + (size_t)dir * EE * HH;

  int w  = threadIdx.x >> 5;
  int mi = w & 1;
  int n0 = blockIdx.x * 64 + (w >> 1) * 16;

  v8f acc = {};
  for (int k0 = 0; k0 < HH; k0 += 32) {
    v16bf a = load_frag(A + (size_t)(mi * 16) * HH + k0, HH);
    v16bf b = load_frag(W + (size_t)n0 * HH + k0, HH);
    acc = wmma_bf16(a, b, acc);
  }

  int l = threadIdx.x & 31;
  int n  = n0 + (l & 15);
  int mb = mi * 16 + ((l >> 4) << 3);
  #pragma unroll
  for (int v = 0; v < 8; v++) {
    int m = mb + v;
    float val = acc[v];
    hp[(size_t)dir * BB * EE + (size_t)m * EE + n] = (bf16)val;
    size_t oi = ((size_t)sOut * BB + m) * EE + n;
    outb[oi] += val;   // safe: fwd/bwd touch different rows; steps serialize
  }
}

// ---------------------------------------------------------------------------
// Host orchestration
// ---------------------------------------------------------------------------
extern "C" void kernel_launch(void* const* d_in, const int* in_sizes, int n_in,
                              void* d_out, int out_size, void* d_ws, size_t ws_size,
                              hipStream_t stream) {
  const float* x    = (const float*)d_in[0];  // (B,S,E)
  const float* W_ih = (const float*)d_in[1];  // (2,2,4096,512)
  const float* W_hh = (const float*)d_in[2];  // (2,2,4096,512)
  const float* W_hr = (const float*)d_in[3];  // (2,2,512,1024)
  const float* bias = (const float*)d_in[4];  // (2,2,4096)
  float* out = (float*)d_out;

  char* ws = (char*)d_ws;
  size_t off = 0;
  auto alloc = [&](size_t bytes) -> void* {
    void* p = ws + off;
    off += (bytes + 255) & ~(size_t)255;
    return p;
  };

  const size_t nXT  = (size_t)SS * BB * EE;          // 8.39M
  const size_t nWih = (size_t)2 * 2 * G4 * EE;       // 8.39M
  const size_t nWhh = nWih;
  const size_t nWhr = (size_t)2 * 2 * EE * HH;       // 2.1M
  const size_t nG   = (size_t)2 * SS * BB * G4;      // 134M floats (537 MB)

  bf16*  xtb  = (bf16*) alloc(nXT * 2);
  bf16*  Wihb = (bf16*) alloc(nWih * 2);
  bf16*  Whhb = (bf16*) alloc(nWhh * 2);
  bf16*  Whrb = (bf16*) alloc(nWhr * 2);
  float* G    = (float*)alloc(nG * 4);
  float* outb = (float*)alloc(nXT * 4);
  bf16*  hb   = (bf16*) alloc((size_t)2 * BB * HH * 2);
  bf16*  hpb  = (bf16*) alloc((size_t)2 * BB * EE * 2);
  float* cbuf = (float*)alloc((size_t)2 * BB * HH * 4);

  const int EW = 4096;  // elementwise grid

  // Stage inputs: transpose + bf16 conversion; outb := xt (layer-0 residual)
  transpose_x_kernel<<<EW, 256, 0, stream>>>(x, outb, xtb);
  f2bf_kernel<<<EW, 256, 0, stream>>>(W_ih, Wihb, nWih);
  f2bf_kernel<<<EW, 256, 0, stream>>>(W_hh, Whhb, nWhh);
  f2bf_kernel<<<EW, 256, 0, stream>>>(W_hr, Whrb, nWhr);

  for (int l = 0; l < 2; l++) {
    // Precompute input-to-gate GEMMs for both directions (34.4 GMAC each)
    for (int dir = 0; dir < 2; dir++) {
      gemm_gates_kernel<<<2048, 256, 0, stream>>>(
          xtb,
          Wihb + (size_t)(l * 2 + dir) * G4 * EE,
          bias + (size_t)(l * 2 + dir) * G4,
          G + (size_t)dir * SS * BB * G4);
    }
    // Reset recurrent state
    zero_f32_kernel<<<64, 256, 0, stream>>>(cbuf, (size_t)2 * BB * HH);
    zero_bf_kernel<<<64, 256, 0, stream>>>(hpb, (size_t)2 * BB * EE);

    // Sequential scan (both directions fused per launch)
    for (int t = 0; t < SS; t++) {
      lstm_gate_step_kernel<<<dim3(HH / 32, 2), 256, 0, stream>>>(
          hpb, Whhb + (size_t)l * 2 * G4 * EE, G, cbuf, hb, t);
      proj_step_kernel<<<dim3(EE / 64, 2), 256, 0, stream>>>(
          hb, Whrb + (size_t)l * 2 * EE * HH, hpb, outb, t);
    }

    if (l == 0) {
      // Layer output (incl. residual) becomes next layer's input; reset accum.
      f2bf_kernel<<<EW, 256, 0, stream>>>(outb, xtb, nXT);
      zero_f32_kernel<<<EW, 256, 0, stream>>>(outb, nXT);
    }
  }

  final_transpose_kernel<<<EW, 256, 0, stream>>>(outb, out);
}
